// CausalMultiHeadSelfAttention_16174846836737
// MI455X (gfx1250) — matrix operand order repaired and checked by reading, not compiled, not run on hardware
//
#include <hip/hip_runtime.h>
#include <hip/hip_bf16.h>

// ---------------------------------------------------------------------------
// CausalMultiHeadSelfAttention for MI455X (gfx1250, wave32, WMMA + TDM).
// b=2, s=2048, d_model=1024, 16 heads x d_k=64. fp32 in/out, f16 WMMA compute
// with fp32 accumulation (v_wmma_f32_16x16x32_f16). V tiles staged to LDS by
// the Tensor Data Mover (tensor_load_to_lds / s_wait_tensorcnt).
// ---------------------------------------------------------------------------

typedef __attribute__((ext_vector_type(16))) _Float16 v16h;
typedef __attribute__((ext_vector_type(8)))  _Float16 v8h;
typedef __attribute__((ext_vector_type(8)))  float    v8f;
typedef unsigned int u32x4 __attribute__((ext_vector_type(4)));
typedef int          i32x4 __attribute__((ext_vector_type(4)));
typedef int          i32x8 __attribute__((ext_vector_type(8)));

#define BATCH   2
#define SEQ     2048
#define DMODEL  1024
#define NHEADS  16
#define DK      64

// ---- WMMA wrapper (probe-confirmed 8-arg signature) ------------------------
__device__ __forceinline__ v8f wmma_f16(v16h a, v16h b, v8f c) {
    return __builtin_amdgcn_wmma_f32_16x16x32_f16(
        /*neg_a=*/false, a, /*neg_b=*/false, b,
        /*c_mod=*/(short)0, c, /*reuse_a=*/false, /*reuse_b=*/false);
}

// ---- A-matrix fragment: 16x32 f16 tile, rows along lanes 0..15 -------------
// lanes 0-15 : elems 0..15 -> K=0..15
// lanes 16-31: elems 0..15 -> K=16..31
__device__ __forceinline__ v16h load_a_f16(const _Float16* __restrict__ base,
                                           int ld, int lane) {
    const int r  = lane & 15;
    const int kh = (lane >> 4) & 1;
    const _Float16* p = base + (size_t)r * ld + kh * 16;
    union { v16h v; v8h h[2]; } u;
    u.h[0] = *(const v8h*)(p);
    u.h[1] = *(const v8h*)(p + 8);
    return u.v;
}

// ---- B-matrix fragment: 32x16 f16 tile given as row-major [N,K] ------------
// lane L holds column N=L&15; K block = (L>=16)*16, 16 contiguous values.
__device__ __forceinline__ v16h load_b_f16(const _Float16* __restrict__ base,
                                           int ld, int lane) {
    const int n  = lane & 15;
    const int kb = (lane >> 4) << 4;
    return *(const v16h*)(base + (size_t)n * ld + kb);
}

// ---------------------------------------------------------------------------
// Tensor Data Mover: async 2D f16 tile load (global -> LDS).
// D# per CDNA5 ISA: Group0 = {count=1, lds_addr, global_addr(57b), type=2};
// Group1 = {data_size=2B, tensor dims, tile dims, row stride}; groups 2/3 = 0
// (2D tensor). Arity differs by toolchain: 5-arg (ROCm 7.2) vs 6-arg
// (therock headers present) -- dispatch on the TDM header's presence.
// ---------------------------------------------------------------------------
#if __has_include(<hip/amd_detail/amd_gfx1250_TDM.h>)
#define TDM_ARITY6 1
#endif

__device__ __forceinline__ void tdm_load_tile_f16(unsigned int lds_off,
                                                  const _Float16* gptr,
                                                  unsigned int tensor_w,
                                                  unsigned int tensor_h,
                                                  unsigned int tile_w,
                                                  unsigned int tile_h,
                                                  unsigned int row_stride) {
    const unsigned long long ga = (unsigned long long)(size_t)gptr;
    u32x4 g0;
    g0[0] = 1u;                                        // count=1, user D#
    g0[1] = lds_off;                                   // lds_addr (bytes)
    g0[2] = (unsigned int)ga;                          // global_addr[31:0]
    g0[3] = (unsigned int)((ga >> 32) & 0x01ffffffu)   // global_addr[56:32]
          | (2u << 30);                                // type=2 ("image")
    i32x8 g1;
    g1[0] = (int)(1u << 16);                           // data_size=1 (2 bytes)
    g1[1] = (int)((tensor_w & 0xffffu) << 16);         // tensor_dim0[15:0]
    g1[2] = (int)((tensor_w >> 16) | ((tensor_h & 0xffffu) << 16));
    g1[3] = (int)((tensor_h >> 16) | (tile_w << 16));  // tile_dim0
    g1[4] = (int)tile_h;                               // tile_dim1 (tile_dim2=0)
    g1[5] = (int)row_stride;                           // tensor_dim0_stride lo32
    g1[6] = 0;
    g1[7] = 0;
    const i32x4 z4 = {0, 0, 0, 0};
#ifdef TDM_ARITY6
    const i32x8 z8 = {0, 0, 0, 0, 0, 0, 0, 0};
    __builtin_amdgcn_tensor_load_to_lds(g0, g1, z4, z4, z8, 0);
#else
    __builtin_amdgcn_tensor_load_to_lds(g0, g1, z4, z4, 0);
#endif
}

// ---------------------------------------------------------------------------
// fp32 -> f16 convert
// ---------------------------------------------------------------------------
__global__ void __launch_bounds__(256)
cvt_f32_to_f16_kernel(const float* __restrict__ in, _Float16* __restrict__ out,
                      int n) {
    int i = blockIdx.x * blockDim.x + threadIdx.x;
    if (i < n) out[i] = (_Float16)in[i];
}

// ---------------------------------------------------------------------------
// WMMA GEMM: C[M,N] = A[M,K] * B[N,K]^T  (both f16 row-major, K contiguous).
// One wave computes a 16x64 C tile: 4 independent accumulators per K-step
// (A fragment reused 4x, 4-deep matrix-pipe ILP). The K loop is unrolled 2x
// as a ping-pong pipeline: stage 0 computes with frag-set 0 while loading
// frag-set 1, stage 1 vice versa -- no register copies between stages.
// mode 0: f16 -> [b,h,s,dk]   mode 1: f16 -> [b,h,dk,s]   mode 2: fp32 [M,N]
// grid = (M/16, N/64). Requires K % 64 == 0.
// ---------------------------------------------------------------------------
__global__ void __launch_bounds__(32)
gemm_wmma_kernel(const _Float16* __restrict__ A, const _Float16* __restrict__ B,
                 void* __restrict__ Cout, int Kd, int N, int mode) {
    const int mt   = blockIdx.x;
    const int nt0  = blockIdx.y * 4;          // first 16-wide col tile
    const int lane = threadIdx.x & 31;

    const _Float16* arow = A + (size_t)(mt * 16) * Kd;
    const _Float16* brow[4];
#pragma unroll
    for (int j = 0; j < 4; ++j)
        brow[j] = B + (size_t)((nt0 + j) * 16) * Kd;

    v8f acc[4] = {};
    // frag-set 0 preloaded with k = 0
    v16h a0 = load_a_f16(arow, Kd, lane);
    v16h b0[4];
#pragma unroll
    for (int j = 0; j < 4; ++j) b0[j] = load_b_f16(brow[j], Kd, lane);

    for (int k0 = 0; k0 < Kd; k0 += 64) {
        // ---- stage 0: load set 1 (k0+32, always valid), compute with set 0
        __builtin_prefetch(arow + (size_t)(lane & 15) * Kd + k0 + 128, 0, 1);
        v16h a1 = load_a_f16(arow + k0 + 32, Kd, lane);
        v16h b1[4];
#pragma unroll
        for (int j = 0; j < 4; ++j)
            b1[j] = load_b_f16(brow[j] + k0 + 32, Kd, lane);
        acc[0] = wmma_f16(a0, b0[0], acc[0]);
        acc[1] = wmma_f16(a0, b0[1], acc[1]);
        acc[2] = wmma_f16(a0, b0[2], acc[2]);
        acc[3] = wmma_f16(a0, b0[3], acc[3]);

        // ---- stage 1: load set 0 (k0+64, if any), compute with set 1 ----
        const bool more = (k0 + 64) < Kd;     // wave-uniform
        if (more) {
            __builtin_prefetch(arow + (size_t)(lane & 15) * Kd + k0 + 160, 0, 1);
            a0 = load_a_f16(arow + k0 + 64, Kd, lane);
#pragma unroll
            for (int j = 0; j < 4; ++j)
                b0[j] = load_b_f16(brow[j] + k0 + 64, Kd, lane);
        }
        acc[0] = wmma_f16(a1, b1[0], acc[0]);
        acc[1] = wmma_f16(a1, b1[1], acc[1]);
        acc[2] = wmma_f16(a1, b1[2], acc[2]);
        acc[3] = wmma_f16(a1, b1[3], acc[3]);
    }

    const int r       = lane & 15;
    const int rowbase = (lane >> 4) << 3;     // 0 or 8
    if (mode == 2) {
        float* C = (float*)Cout;
#pragma unroll
        for (int j = 0; j < 4; ++j) {
            const int n = (nt0 + j) * 16 + r;
#pragma unroll
            for (int v = 0; v < 8; ++v) {
                int m = mt * 16 + rowbase + v;
                C[(size_t)m * N + n] = acc[j][v];
            }
        }
    } else {
        _Float16* C = (_Float16*)Cout;
#pragma unroll
        for (int j = 0; j < 4; ++j) {
            const int n  = (nt0 + j) * 16 + r;
            const int hh = n >> 6;            // head
            const int dd = n & 63;            // within-head dim
#pragma unroll
            for (int v = 0; v < 8; ++v) {
                int m  = mt * 16 + rowbase + v;
                int bb = m >> 11;             // batch
                int ss = m & 2047;            // seq pos
                size_t idx;
                if (mode == 0)                // [b,h,s,dk]
                    idx = ((((size_t)bb * NHEADS + hh) * SEQ + ss) * DK) + dd;
                else                          // [b,h,dk,s]
                    idx = ((((size_t)bb * NHEADS + hh) * DK + dd) * SEQ) + ss;
                C[idx] = (_Float16)acc[j][v];
            }
        }
    }
}

// ---------------------------------------------------------------------------
// RoPE on Q and K ([b,h,s,dk] f16, in place). Q additionally scaled by 1/8
// (= 1/sqrt(dk)), folding the score scale into Q.
// ---------------------------------------------------------------------------
__global__ void __launch_bounds__(256)
rope_kernel(_Float16* __restrict__ Q, _Float16* __restrict__ K,
            const int* __restrict__ pos, int npairs) {
    int p = blockIdx.x * blockDim.x + threadIdx.x;
    if (p >= npairs) return;
    const int dd2 = p & (DK / 2 - 1);         // pair index 0..31
    const int ss  = (p >> 5) & (SEQ - 1);     // seq pos
    const float fpos = (float)pos[ss];
    const float inv = __expf((float)dd2 * (-2.0f / (float)DK) * 9.21034037198f);
    const float ang = fpos * inv;
    float sn, cs;
    __sincosf(ang, &sn, &cs);
    const size_t i0 = (size_t)p * 2;
    {
        float q1 = (float)Q[i0], q2 = (float)Q[i0 + 1];
        Q[i0]     = (_Float16)((q1 * cs - q2 * sn) * 0.125f);
        Q[i0 + 1] = (_Float16)((q1 * sn + q2 * cs) * 0.125f);
    }
    {
        float k1 = (float)K[i0], k2 = (float)K[i0 + 1];
        K[i0]     = (_Float16)(k1 * cs - k2 * sn);
        K[i0 + 1] = (_Float16)(k1 * sn + k2 * cs);
    }
}

// ---------------------------------------------------------------------------
// Causal attention, one block (4 waves) per (qtile, head, batch).
// LDS: 16x2048 fp32 score strip (128 KB) + per-wave double-buffered V tiles
// (8 KB) staged by the Tensor Data Mover. Softmax in-LDS; unnormalized exp()
// overlaid as an f16 P-strip; normalization deferred to the epilogue.
// ---------------------------------------------------------------------------
__global__ void __launch_bounds__(128)
attn_kernel(const _Float16* __restrict__ Q, const _Float16* __restrict__ K,
            const _Float16* __restrict__ VT, _Float16* __restrict__ attn_out) {
    extern __shared__ char smem[];
    float* sS = (float*)smem;                    // [16][SEQ] fp32 scores
    _Float16* sV = (_Float16*)(smem + 16 * SEQ * sizeof(float)); // 4w x 2 x 512
    __shared__ float sInv[16];                   // per-row 1/sum

    const int qt   = blockIdx.x;                 // 0..127
    const int h    = blockIdx.y;
    const int b    = blockIdx.z;
    const int wave = threadIdx.x >> 5;           // 0..3
    const int lane = threadIdx.x & 31;
    const int r    = lane & 15;

    const size_t head_off = (size_t)(b * NHEADS + h) * SEQ * DK;
    const _Float16* Qh = Q  + head_off;          // [s][dk]
    const _Float16* Kh = K  + head_off;          // [s][dk]
    const _Float16* Vt = VT + head_off;          // [dk][s]

    // Q A-fragments for rows qt*16..qt*16+15, d = 0..31 and 32..63
    const v16h qa0 = load_a_f16(Qh + (size_t)(qt * 16) * DK + 0,  DK, lane);
    const v16h qa1 = load_a_f16(Qh + (size_t)(qt * 16) * DK + 32, DK, lane);

    // ---- Phase 1: scores = Q K^T (causal tiles only) ----
    for (int kt = wave; kt <= qt; kt += 4) {
        v16h kb0 = load_b_f16(Kh + (size_t)(kt * 16) * DK + 0,  DK, lane);
        v16h kb1 = load_b_f16(Kh + (size_t)(kt * 16) * DK + 32, DK, lane);
        v8f c = {};
        c = wmma_f16(qa0, kb0, c);
        c = wmma_f16(qa1, kb1, c);
        const int colg    = kt * 16 + r;
        const int rowbase = (lane >> 4) << 3;
#pragma unroll
        for (int v = 0; v < 8; ++v) {
            int row  = rowbase + v;
            int rowg = qt * 16 + row;
            sS[(size_t)row * SEQ + colg] =
                (colg <= rowg) ? c[v] : -__builtin_inff();
        }
    }
    __syncthreads();

    const int kend   = (qt + 1) * 16;
    const int kend32 = (kend + 31) & ~31;

    // ---- Phase 2: row softmax; write unnormalized exp() as f16 P-strip ----
    for (int row = wave; row < 16; row += 4) {
        const int qg  = qt * 16 + row;
        const int len = qg + 1;
        float* srow = sS + (size_t)row * SEQ;
        _Float16* prow = (_Float16*)srow;        // f16 overlay, same row base
        float mx = -__builtin_inff();
        for (int c0 = lane; c0 < len; c0 += 32) mx = fmaxf(mx, srow[c0]);
        for (int off = 16; off >= 1; off >>= 1)
            mx = fmaxf(mx, __shfl_xor(mx, off, 32));
        float sum = 0.f;
        for (int c0 = lane; c0 < kend32; c0 += 32) {
            float e = (c0 < len) ? __expf(srow[c0] - mx) : 0.f;
            sum += e;
            prow[c0] = (_Float16)e;              // writes trail reads (lockstep)
        }
        for (int off = 16; off >= 1; off >>= 1) sum += __shfl_xor(sum, off, 32);
        if (lane == 0) sInv[row] = 1.0f / sum;
    }
    __syncthreads();

    // ---- Phase 3: out = P @ V ; wave w owns output cols d = 16w..16w+15 ----
    // V^T tiles (16 rows x 32 f16, row pitch SEQ) are DMA'd into LDS by the
    // TDM, double-buffered; compute overlaps the next tile's transfer.
    const _Float16* sP = (const _Float16*)sS;    // row pitch 4096 f16
    const _Float16* vrow = Vt + (size_t)(wave * 16) * SEQ;
    _Float16* vbuf = sV + wave * (2 * 16 * 32);  // two 512-f16 buffers
    const unsigned int vbuf_off = (unsigned int)(size_t)vbuf;

    tdm_load_tile_f16(vbuf_off, vrow, SEQ, DK, 32, 16, SEQ);
    int buf = 0;
    v8f acc = {};
    for (int k0 = 0; k0 < kend32; k0 += 32) {
        const bool more = (k0 + 32) < kend32;    // wave-uniform
        if (more) {
            tdm_load_tile_f16(vbuf_off + (buf ^ 1) * 1024, vrow + k0 + 32,
                              SEQ, DK, 32, 16, SEQ);
            __builtin_amdgcn_s_wait_tensorcnt(1);  // current tile landed
        } else {
            __builtin_amdgcn_s_wait_tensorcnt(0);  // drain
        }
        v16h pa = load_a_f16(sP + k0, 2 * SEQ, lane);     // LDS, ld = 4096
        v16h vb = load_b_f16(vbuf + buf * 512, 32, lane); // LDS, ld = 32
        acc = wmma_f16(pa, vb, acc);
        buf ^= 1;
    }
    const int coln    = h * DK + wave * 16 + r;  // column in [b,s,1024]
    const int rowbase = (lane >> 4) << 3;
#pragma unroll
    for (int v = 0; v < 8; ++v) {
        int rowg = qt * 16 + rowbase + v;
        float o  = acc[v] * sInv[rowbase + v];   // deferred softmax normalize
        attn_out[((size_t)(b * SEQ + rowg)) * DMODEL + coln] = (_Float16)o;
    }
}

// ---------------------------------------------------------------------------
// Host launcher
// ---------------------------------------------------------------------------
extern "C" void kernel_launch(void* const* d_in, const int* in_sizes, int n_in,
                              void* d_out, int out_size, void* d_ws, size_t ws_size,
                              hipStream_t stream) {
    const float* x  = (const float*)d_in[0];   // [2,2048,1024]
    const int*   tp = (const int*)  d_in[1];   // [2048]
    const float* Wq = (const float*)d_in[2];   // [1024,1024]
    const float* Wk = (const float*)d_in[3];
    const float* Wv = (const float*)d_in[4];
    const float* Wo = (const float*)d_in[5];
    float* out = (float*)d_out;                // [2,2048,1024]

    // ---- workspace partition (all f16, 48 MB total) ----
    char* ws = (char*)d_ws;
    const size_t SZ_X  = (size_t)BATCH * SEQ * DMODEL * sizeof(_Float16); // 8 MB
    const size_t SZ_W  = (size_t)DMODEL * DMODEL * sizeof(_Float16);      // 2 MB
    _Float16* xh   = (_Float16*)(ws);
    _Float16* wqh  = (_Float16*)(ws + SZ_X);
    _Float16* wkh  = (_Float16*)(ws + SZ_X + 1 * SZ_W);
    _Float16* wvh  = (_Float16*)(ws + SZ_X + 2 * SZ_W);
    _Float16* woh  = (_Float16*)(ws + SZ_X + 3 * SZ_W);
    _Float16* qrw  = (_Float16*)(ws + SZ_X + 4 * SZ_W);            // [b,h,s,dk]
    _Float16* krw  = (_Float16*)(ws + 2 * SZ_X + 4 * SZ_W);        // [b,h,s,dk]
    _Float16* vt   = (_Float16*)(ws + 3 * SZ_X + 4 * SZ_W);        // [b,h,dk,s]
    _Float16* ath  = (_Float16*)(ws + 4 * SZ_X + 4 * SZ_W);        // [b,s,1024]

    const int M  = BATCH * SEQ;  // 4096
    const int NX = BATCH * SEQ * DMODEL;
    const int NW = DMODEL * DMODEL;

    // 1) fp32 -> f16 conversions
    cvt_f32_to_f16_kernel<<<(NX + 255) / 256, 256, 0, stream>>>(x,  xh,  NX);
    cvt_f32_to_f16_kernel<<<(NW + 255) / 256, 256, 0, stream>>>(Wq, wqh, NW);
    cvt_f32_to_f16_kernel<<<(NW + 255) / 256, 256, 0, stream>>>(Wk, wkh, NW);
    cvt_f32_to_f16_kernel<<<(NW + 255) / 256, 256, 0, stream>>>(Wv, wvh, NW);
    cvt_f32_to_f16_kernel<<<(NW + 255) / 256, 256, 0, stream>>>(Wo, woh, NW);

    // 2) projections: one wave per 16x64 C tile
    dim3 gProj(M / 16, DMODEL / 64);
    gemm_wmma_kernel<<<gProj, 32, 0, stream>>>(xh, wqh, qrw, DMODEL, DMODEL, 0);
    gemm_wmma_kernel<<<gProj, 32, 0, stream>>>(xh, wkh, krw, DMODEL, DMODEL, 0);
    gemm_wmma_kernel<<<gProj, 32, 0, stream>>>(xh, wvh, vt,  DMODEL, DMODEL, 1);

    // 3) RoPE on Q (scaled by 1/sqrt(dk)) and K
    const int npairs = BATCH * NHEADS * SEQ * (DK / 2);  // 2,097,152
    rope_kernel<<<(npairs + 255) / 256, 256, 0, stream>>>(qrw, krw, tp, npairs);

    // 4) causal attention: block per (qtile, head, batch);
    //    LDS = 128 KB score strip + 8 KB TDM V-tile double buffers
    dim3 gAttn(SEQ / 16, NHEADS, BATCH);
    const size_t smem = 16 * SEQ * sizeof(float) + 4 * 2 * 16 * 32 * sizeof(_Float16);
    attn_kernel<<<gAttn, 128, smem, stream>>>(qrw, krw, vt, ath);

    // 5) output projection: out = attn @ Wo^T, fp32 result
    gemm_wmma_kernel<<<gProj, 32, 0, stream>>>(ath, woh, out, DMODEL, DMODEL, 2);
}
